// IntegratedModelV2_12618613915700
// MI455X (gfx1250) — compile-verified, matrix operand order checked by reading, not run on hardware
//
#include <hip/hip_runtime.h>
#include <hip/hip_bf16.h>
#include <math.h>

// ---------------- WMMA fragment types (CDNA5 gfx1250, wave32) ----------------
typedef __attribute__((ext_vector_type(16))) __bf16 v16bf;
typedef __attribute__((ext_vector_type(8)))  float  v8f;
typedef __attribute__((ext_vector_type(4)))  unsigned int v4u;
typedef __attribute__((ext_vector_type(8)))  int v8i_t;
typedef __attribute__((ext_vector_type(4)))  int v4i_t;

union Frag { v16bf v; unsigned short u[16]; };

__device__ inline unsigned short f2bf(float f) {
    unsigned int x = __float_as_uint(f);
    x += 0x7fffu + ((x >> 16) & 1u);           // round-to-nearest-even
    return (unsigned short)(x >> 16);
}

// Fragment gather per ISA 7.12.2 (16-bit A 16x32): lane<16 holds K=hi8+0..7 and
// hi8+16..23, hi8 = (lane>=16)*8.  B stored [N][K] uses the identical pattern.
__device__ inline void load_frag(Frag& f, const unsigned short* row, int hi8) {
#pragma unroll
    for (int i = 0; i < 8; i++) {
        f.u[i]     = row[hi8 + i];
        f.u[i + 8] = row[hi8 + 16 + i];
    }
}

__device__ inline float gelu_exact(float v) {
    return 0.5f * v * (1.0f + erff(v * 0.7071067811865475f));
}

// ---------------- Tensor Data Mover staging (CDNA5) ----------------
#if defined(__gfx1250__) && __has_builtin(__builtin_amdgcn_tensor_load_to_lds) && __has_builtin(__builtin_amdgcn_s_wait_tensorcnt)
#define USE_TDM 1
#else
#define USE_TDM 0
#endif

__device__ inline unsigned lds_addr_of(const void* p) {
    // generic LDS address: low 32 bits == LDS byte offset (aperture rule, ISA 10.2)
    return (unsigned)(unsigned long long)p;
}

// Load a 2-D tile: `rows` rows x 32 bf16 elements, row stride K elements,
// into contiguous LDS (rows*64 bytes).  D# per ISA 8.3/8.4.
__device__ inline void tdm_load_tile(const unsigned short* gsrc, unsigned lds_off,
                                     unsigned K, unsigned rows) {
#if USE_TDM
    unsigned long long ga = (unsigned long long)gsrc;
    const unsigned TD1 = 1u << 20;                 // tensor_dim1 (never OOB)
    v4u g0;
    g0[0] = 1u;                                    // count=1, user desc
    g0[1] = lds_off;                               // lds_addr
    g0[2] = (unsigned)(ga & 0xffffffffull);        // global_addr[31:0]
    g0[3] = (unsigned)((ga >> 32) & 0x01ffffffull) | (2u << 30); // addr[56:32], type=2
    v8i_t g1;
    g1[0] = (int)(1u << 16);                                   // data_size=1 (2 bytes)
    g1[1] = (int)((K & 0xffffu) << 16);                        // tensor_dim0[15:0]
    g1[2] = (int)(((K >> 16) & 0xffffu) | ((TD1 & 0xffffu) << 16));
    g1[3] = (int)(((TD1 >> 16) & 0xffffu) | (32u << 16));      // tile_dim0=32
    g1[4] = (int)rows;                                         // tile_dim1, tile_dim2=0
    g1[5] = (int)K;                                            // tensor_dim0_stride lo
    g1[6] = 0;
    g1[7] = 0;
    v4i_t z4 = {0, 0, 0, 0};
    v8i_t z8 = {0, 0, 0, 0, 0, 0, 0, 0};
    __builtin_amdgcn_tensor_load_to_lds(g0, g1, z4, z4, z8, 0);
#else
    (void)gsrc; (void)lds_off; (void)K; (void)rows;
#endif
}

// Manual fallback / non-TDM staging: 128x32 u16 via 16B chunks, 256 threads.
__device__ inline void stage_manual(const unsigned short* __restrict__ G, int K,
                                    unsigned short (*dst)[32], int tid) {
    for (int id = tid; id < 512; id += 256) {
        int r = id >> 2, c4 = id & 3;
        *(uint4*)&dst[r][c4 * 8] = *(const uint4*)(G + (size_t)r * K + c4 * 8);
    }
}

// ---------------- WMMA GEMM:  C[M,N] = act(A[M,K] @ W[N,K]^T + bias) ----------------
// A,W bf16.  BM=BN=128, BK=32, 256 threads (8 waves).  Wave owns 32x64 of C:
// 2 A-frags x 4 B-frags -> 8 v_wmma per K step.  Double-buffered LDS staged by TDM.
// Requires M%128==0, N%128==0, K%32==0 (true for every call site).
__global__ __launch_bounds__(256) void gemm_bf16_k(
    const unsigned short* __restrict__ A, const unsigned short* __restrict__ W,
    const float* __restrict__ bias, int has_bias,
    float* __restrict__ Cf, unsigned short* __restrict__ Cb, int out_bf16,
    int M, int N, int K, int act)
{
    __shared__ __align__(16) unsigned short As[2][128][32];
    __shared__ __align__(16) unsigned short Bs[2][128][32];

    const int tid = threadIdx.x, lane = tid & 31, wave = tid >> 5;
    const int bm = blockIdx.x * 128, bn = blockIdx.y * 128;
    const int wm = (wave & 3) * 32;        // 4 wave-rows cover 128 rows
    const int wn = (wave >> 2) * 64;       // 2 wave-cols cover 128 cols
    const int ml = lane & 15;
    const int hi8 = (lane >> 4) << 3;

    const unsigned short* Abase = A + (size_t)bm * K;
    const unsigned short* Wbase = W + (size_t)bn * K;

    v8f acc[2][4];
#pragma unroll
    for (int i = 0; i < 2; i++)
#pragma unroll
        for (int j = 0; j < 4; j++)
            acc[i][j] = (v8f){0.f,0.f,0.f,0.f,0.f,0.f,0.f,0.f};

#if USE_TDM
    if (wave == 0)      tdm_load_tile(Abase, lds_addr_of(&As[0][0][0]), (unsigned)K, 128);
    else if (wave == 1) tdm_load_tile(Wbase, lds_addr_of(&Bs[0][0][0]), (unsigned)K, 128);
    if (wave < 2) __builtin_amdgcn_s_wait_tensorcnt(0);
#else
    stage_manual(Abase, K, As[0], tid);
    stage_manual(Wbase, K, Bs[0], tid);
#endif
    __syncthreads();

    int cur = 0;
    for (int k0 = 0; k0 < K; k0 += 32) {
        const int nxt = cur ^ 1;
        const bool more = (k0 + 32) < K;
        if (more) {
#if USE_TDM
            if (wave == 0)      tdm_load_tile(Abase + k0 + 32, lds_addr_of(&As[nxt][0][0]), (unsigned)K, 128);
            else if (wave == 1) tdm_load_tile(Wbase + k0 + 32, lds_addr_of(&Bs[nxt][0][0]), (unsigned)K, 128);
#else
            stage_manual(Abase + k0 + 32, K, As[nxt], tid);
            stage_manual(Wbase + k0 + 32, K, Bs[nxt], tid);
#endif
        }
        Frag a0, a1, b0, b1, b2, b3;
        load_frag(a0, As[cur][wm + ml],      hi8);
        load_frag(a1, As[cur][wm + 16 + ml], hi8);
        load_frag(b0, Bs[cur][wn + ml],      hi8);
        load_frag(b1, Bs[cur][wn + 16 + ml], hi8);
        load_frag(b2, Bs[cur][wn + 32 + ml], hi8);
        load_frag(b3, Bs[cur][wn + 48 + ml], hi8);
        acc[0][0] = __builtin_amdgcn_wmma_f32_16x16x32_bf16(false, a0.v, false, b0.v, (short)0, acc[0][0], false, false);
        acc[0][1] = __builtin_amdgcn_wmma_f32_16x16x32_bf16(false, a0.v, false, b1.v, (short)0, acc[0][1], false, false);
        acc[0][2] = __builtin_amdgcn_wmma_f32_16x16x32_bf16(false, a0.v, false, b2.v, (short)0, acc[0][2], false, false);
        acc[0][3] = __builtin_amdgcn_wmma_f32_16x16x32_bf16(false, a0.v, false, b3.v, (short)0, acc[0][3], false, false);
        acc[1][0] = __builtin_amdgcn_wmma_f32_16x16x32_bf16(false, a1.v, false, b0.v, (short)0, acc[1][0], false, false);
        acc[1][1] = __builtin_amdgcn_wmma_f32_16x16x32_bf16(false, a1.v, false, b1.v, (short)0, acc[1][1], false, false);
        acc[1][2] = __builtin_amdgcn_wmma_f32_16x16x32_bf16(false, a1.v, false, b2.v, (short)0, acc[1][2], false, false);
        acc[1][3] = __builtin_amdgcn_wmma_f32_16x16x32_bf16(false, a1.v, false, b3.v, (short)0, acc[1][3], false, false);
#if USE_TDM
        if (more && wave < 2) __builtin_amdgcn_s_wait_tensorcnt(0);
#endif
        __syncthreads();
        cur = nxt;
    }

    const int moff = (lane & 16) ? 8 : 0;
#pragma unroll
    for (int ta = 0; ta < 2; ta++) {
#pragma unroll
        for (int tn = 0; tn < 4; tn++) {
            const int gn = bn + wn + tn * 16 + ml;
            const float bv = has_bias ? bias[gn] : 0.f;
#pragma unroll
            for (int r = 0; r < 8; r++) {
                int gm = bm + wm + ta * 16 + moff + r;
                float v = acc[ta][tn][r] + bv;
                if (act == 1) v = gelu_exact(v);
                if (out_bf16) Cb[(size_t)gm * N + gn] = f2bf(v);
                else          Cf[(size_t)gm * N + gn] = v;
            }
        }
    }
}

// ---------------- Windowed attention, one block per (window, head), hd=32 ----------------
// qkv bf16 [T,3C] in raw token order; gathers with cyclic shift, writes bf16 [T,C].
__global__ __launch_bounds__(256) void win_attn_k(
    const unsigned short* __restrict__ qkv, const float* __restrict__ rpb,
    unsigned short* __restrict__ out, int Hs, int C, int heads, int shift)
{
    __shared__ __align__(16) unsigned short qs[64][32];
    __shared__ __align__(16) unsigned short ks[64][32];
    __shared__ __align__(16) unsigned short vsT[32][64];
    __shared__ float          S[64][64];
    __shared__ __align__(16) unsigned short Ps[64][64];

    const int nw   = Hs / 7;
    const int win  = blockIdx.x % (nw * nw);
    const int b    = blockIdx.x / (nw * nw);
    const int head = blockIdx.y;
    const int wy = win / nw, wx = win % nw;
    const int tid = threadIdx.x, lane = tid & 31, wave = tid >> 5;
    const int ml  = lane & 15;
    const int hi8 = (lane >> 4) << 3;

    for (int idx = tid; idx < 64 * 32; idx += 256) {
        int t = idx >> 5, d = idx & 31;
        unsigned short qv = 0, kv = 0, vv = 0;
        if (t < 49) {
            int i = t / 7, j = t % 7;
            int y = (wy * 7 + i + shift) % Hs;
            int x = (wx * 7 + j + shift) % Hs;
            size_t tok = ((size_t)b * Hs + y) * Hs + x;
            const unsigned short* base = qkv + tok * (size_t)(3 * C) + head * 32 + d;
            qv = base[0]; kv = base[C]; vv = base[2 * C];
        }
        qs[t][d] = qv; ks[t][d] = kv; vsT[d][t] = vv;
    }
    __syncthreads();

    // S = q @ k^T : 16 tiles (K=32 -> one WMMA each), 2 per wave
    for (int t = wave; t < 16; t += 8) {
        int tm = (t >> 2) * 16, tn = (t & 3) * 16;
        Frag a, bf;
        load_frag(a,  qs[tm + ml], hi8);
        load_frag(bf, ks[tn + ml], hi8);
        v8f c = {0.f,0.f,0.f,0.f,0.f,0.f,0.f,0.f};
        c = __builtin_amdgcn_wmma_f32_16x16x32_bf16(false, a.v, false, bf.v, (short)0, c, false, false);
        int rb = tm + ((lane & 16) ? 8 : 0);
#pragma unroll
        for (int r = 0; r < 8; r++) S[rb + r][tn + ml] = c[r];
    }
    __syncthreads();

    // scale + rel-pos bias (+ shift mask) + softmax over 49 keys
    if (tid < 64) {
        int mrow = tid;
        if (mrow < 49) {
            const float scale = 0.17677669529663687f; // 1/sqrt(32)
            int mi = mrow / 7, mj = mrow % 7;
            int mlab = 0;
            if (shift > 0) {
                int ry = wy * 7 + mi, rx = wx * 7 + mj;
                int ly = (ry < Hs - 7) ? 0 : ((ry < Hs - shift) ? 1 : 2);
                int lx = (rx < Hs - 7) ? 0 : ((rx < Hs - shift) ? 1 : 2);
                mlab = ly * 3 + lx;
            }
            float mx = -1e30f;
            for (int n = 0; n < 49; n++) {
                int ni = n / 7, nj = n % 7;
                int rel = (mi - ni + 6) * 13 + (mj - nj + 6);
                float v = S[mrow][n] * scale + rpb[rel * heads + head];
                if (shift > 0) {
                    int ry = wy * 7 + ni, rx = wx * 7 + nj;
                    int ly = (ry < Hs - 7) ? 0 : ((ry < Hs - shift) ? 1 : 2);
                    int lx = (rx < Hs - 7) ? 0 : ((rx < Hs - shift) ? 1 : 2);
                    if (ly * 3 + lx != mlab) v += -100.f;
                }
                S[mrow][n] = v;
                mx = fmaxf(mx, v);
            }
            float sum = 0.f;
            for (int n = 0; n < 49; n++) { float e = __expf(S[mrow][n] - mx); S[mrow][n] = e; sum += e; }
            float inv = 1.f / sum;
            for (int n = 0; n < 49; n++) Ps[mrow][n] = f2bf(S[mrow][n] * inv);
            for (int n = 49; n < 64; n++) Ps[mrow][n] = 0;
        } else {
            for (int n = 0; n < 64; n++) Ps[mrow][n] = 0;
        }
    }
    __syncthreads();

    // O = P @ V : 4 m-tiles x 2 n-tiles, one per wave, K=64 (2 WMMA)
    {
        int tm = (wave & 3) * 16, tn = (wave >> 2) * 16;
        v8f acc = {0.f,0.f,0.f,0.f,0.f,0.f,0.f,0.f};
#pragma unroll
        for (int kb = 0; kb < 64; kb += 32) {
            Frag a, bf;
            load_frag(a,  &Ps[tm + ml][kb],  hi8);
            load_frag(bf, &vsT[tn + ml][kb], hi8);
            acc = __builtin_amdgcn_wmma_f32_16x16x32_bf16(false, a.v, false, bf.v, (short)0, acc, false, false);
        }
        int rb = tm + ((lane & 16) ? 8 : 0);
#pragma unroll
        for (int r = 0; r < 8; r++) {
            int mrow = rb + r;
            if (mrow < 49) {
                int i = mrow / 7, j = mrow % 7;
                int y = (wy * 7 + i + shift) % Hs;
                int x = (wx * 7 + j + shift) % Hs;
                size_t tok = ((size_t)b * Hs + y) * Hs + x;
                out[tok * (size_t)C + head * 32 + tn + ml] = f2bf(acc[r]);
            }
        }
    }
}

// ---------------- Scalar helper kernels ----------------
__global__ void cvt_bf16_k(const float* __restrict__ in, unsigned short* __restrict__ out, size_t n)
{
    size_t i = (size_t)blockIdx.x * blockDim.x + threadIdx.x;
    size_t stride = (size_t)gridDim.x * blockDim.x;
    for (; i < n; i += stride) out[i] = f2bf(in[i]);
}

__global__ void conv1_bn_relu_k(const float* __restrict__ x, const float* __restrict__ w,
                                const float* __restrict__ bias,
                                const float* __restrict__ g, const float* __restrict__ bb,
                                const float* __restrict__ mu, const float* __restrict__ var,
                                float* __restrict__ y)
{
    int idx = blockIdx.x * blockDim.x + threadIdx.x;
    const int HW = 224 * 224, total = 32 * 3 * HW;
    if (idx >= total) return;
    int px = idx % 224, py = (idx / 224) % 224, co = (idx / HW) % 3, b = idx / (3 * HW);
    float acc = bias[co];
    for (int ci = 0; ci < 3; ci++)
        for (int ky = 0; ky < 3; ky++)
            for (int kx = 0; kx < 3; kx++) {
                int iy = py + ky - 1, ix = px + kx - 1;
                if (iy >= 0 && iy < 224 && ix >= 0 && ix < 224)
                    acc += x[((size_t)(b * 3 + ci) * 224 + iy) * 224 + ix] * w[((co * 3 + ci) * 3 + ky) * 3 + kx];
            }
    acc = (acc - mu[co]) * rsqrtf(var[co] + 1e-5f);
    acc = acc * g[co] + bb[co];
    y[idx] = fmaxf(acc, 0.f);
}

__global__ void edge_mul_k(const float* __restrict__ h, const float* __restrict__ w,
                           const float* __restrict__ b2, float* __restrict__ y)
{
    int idx = blockIdx.x * blockDim.x + threadIdx.x;
    const int HW = 224 * 224, total = 32 * HW;
    if (idx >= total) return;
    int p = idx % HW, b = idx / HW;
    float e = b2[0];
    for (int c = 0; c < 3; c++) e += h[((size_t)b * 3 + c) * HW + p] * w[c];
    e = 1.f / (1.f + __expf(-e));
    for (int c = 0; c < 3; c++) {
        size_t o = ((size_t)b * 3 + c) * HW + p;
        y[o] = h[o] * e;
    }
}

__global__ void patch_embed_k(const float* __restrict__ x, const float* __restrict__ w,
                              const float* __restrict__ bias, float* __restrict__ y)
{
    int idx = blockIdx.x * blockDim.x + threadIdx.x;
    const int total = 32 * 56 * 56 * 128;
    if (idx >= total) return;
    int e = idx % 128, t = idx / 128;
    int xq = t % 56, yq = (t / 56) % 56, b = t / (56 * 56);
    float acc = bias[e];
    for (int c = 0; c < 3; c++)
        for (int ky = 0; ky < 4; ky++)
            for (int kx = 0; kx < 4; kx++)
                acc += x[((size_t)(b * 3 + c) * 224 + yq * 4 + ky) * 224 + xq * 4 + kx] *
                       w[((e * 3 + c) * 4 + ky) * 4 + kx];
    y[idx] = acc;
}

// LayerNorm: fp32 in, bf16 out (feeds WMMA GEMMs). One wave per token.
__global__ __launch_bounds__(256) void layernorm_k(const float* __restrict__ x,
                                                   const float* __restrict__ g, const float* __restrict__ b,
                                                   unsigned short* __restrict__ y, int T, int C)
{
    int wave = threadIdx.x >> 5, lane = threadIdx.x & 31;
    int t = blockIdx.x * 8 + wave;
    if (t >= T) return;
    const float* xr = x + (size_t)t * C;
    float s = 0.f, s2 = 0.f;
    for (int c = lane; c < C; c += 32) { float v = xr[c]; s += v; s2 += v * v; }
#pragma unroll
    for (int o = 16; o > 0; o >>= 1) { s += __shfl_xor(s, o, 32); s2 += __shfl_xor(s2, o, 32); }
    float mu = s / C;
    float var = s2 / C - mu * mu;
    float inv = rsqrtf(var + 1e-5f);
    unsigned short* yr = y + (size_t)t * C;
    for (int c = lane; c < C; c += 32) yr[c] = f2bf((xr[c] - mu) * inv * g[c] + b[c]);
}

__global__ void add_inplace_k(float* __restrict__ y, const float* __restrict__ a, size_t n)
{
    size_t i = (size_t)blockIdx.x * blockDim.x + threadIdx.x;
    if (i < n) y[i] += a[i];
}

__global__ void pm_gather_k(const float* __restrict__ x, float* __restrict__ y, int H2, int C)
{
    size_t idx = (size_t)blockIdx.x * blockDim.x + threadIdx.x;
    size_t total = (size_t)32 * H2 * H2 * 4 * C;
    if (idx >= total) return;
    int cc = idx % (4 * C);
    size_t t = idx / (4 * C);
    int x2 = t % H2, y2 = (t / H2) % H2, b = t / ((size_t)H2 * H2);
    int q = cc / C, c = cc % C;
    int dy = (q == 1 || q == 3) ? 1 : 0;
    int dx = (q >= 2) ? 1 : 0;
    y[idx] = x[(((size_t)b * (2 * H2) + (2 * y2 + dy)) * (2 * H2) + (2 * x2 + dx)) * C + c];
}

__global__ void meanpool_k(const float* __restrict__ x, float* __restrict__ y, int H, int C)
{
    int idx = blockIdx.x * blockDim.x + threadIdx.x;
    if (idx >= 32 * C) return;
    int c = idx % C, b = idx / C;
    float s = 0.f;
    for (int p = 0; p < H * H; p++) s += x[((size_t)b * H * H + p) * C + c];
    y[idx] = s / (float)(H * H);
}

__global__ void dense_elu_k(const float* __restrict__ in, const float* __restrict__ w,
                            const float* __restrict__ bias, float* __restrict__ out, int N, int K)
{
    int idx = blockIdx.x * blockDim.x + threadIdx.x;
    if (idx >= 32 * N) return;
    int n = idx % N, b = idx / N;
    float acc = bias[n];
    for (int k = 0; k < K; k++) acc += in[(size_t)b * K + k] * w[(size_t)n * K + k];
    out[idx] = acc > 0.f ? acc : (__expf(acc) - 1.f);
}

__global__ void dense_out_k(const float* __restrict__ h, const float* __restrict__ w,
                            const float* __restrict__ bias, float* __restrict__ out)
{
    int b = threadIdx.x;
    if (b >= 32) return;
    float acc = bias[0];
    for (int k = 0; k < 512; k++) acc += h[(size_t)b * 512 + k] * w[k];
    out[b] = acc;
}

// ---------------- Host orchestration ----------------
static inline void gemm(const unsigned short* A, const unsigned short* W, const float* bias,
                        int has_bias, float* Cf, unsigned short* Cb, int out_bf16,
                        int M, int N, int K, int act, hipStream_t s)
{
    dim3 g(M / 128, N / 128);
    gemm_bf16_k<<<g, 256, 0, s>>>(A, W, bias, has_bias, Cf, Cb, out_bf16, M, N, K, act);
}

static void run_stage(void* const* d_in, int base,
                      const unsigned short* qkvW, const unsigned short* projW,
                      const unsigned short* fc1W, const unsigned short* fc2W,
                      float* X, float* TMPF, unsigned short* LNB,
                      unsigned short* ATNB, unsigned short* BIGB,
                      int T, int Hs, int C, int heads, int nb, hipStream_t s)
{
    const float* ln1_g = (const float*)d_in[base + 0];
    const float* ln1_b = (const float*)d_in[base + 1];
    const float* qkv_b = (const float*)d_in[base + 3];
    const float* rpb   = (const float*)d_in[base + 4];
    const float* prj_b = (const float*)d_in[base + 6];
    const float* ln2_g = (const float*)d_in[base + 7];
    const float* ln2_b = (const float*)d_in[base + 8];
    const float* fc1_b = (const float*)d_in[base + 10];
    const float* fc2_b = (const float*)d_in[base + 12];

    const int C3 = 3 * C, C4 = 4 * C;
    const int nw = Hs / 7;
    const size_t TC = (size_t)T * C;

    for (int i = 0; i < nb; i++) {
        int shift = (i % 2 == 0) ? 0 : 3;
        const float* g1 = ln1_g + (size_t)i * C;
        const float* b1 = ln1_b + (size_t)i * C;
        const unsigned short* qw = qkvW + (size_t)i * C3 * C;
        const float* qb = qkv_b + (size_t)i * C3;
        const float* rp = rpb + (size_t)i * 169 * heads;
        const unsigned short* pw = projW + (size_t)i * C * C;
        const float* pb = prj_b + (size_t)i * C;
        const float* g2 = ln2_g + (size_t)i * C;
        const float* b2 = ln2_b + (size_t)i * C;
        const unsigned short* f1w = fc1W + (size_t)i * C4 * C;
        const float* f1b = fc1_b + (size_t)i * C4;
        const unsigned short* f2w = fc2W + (size_t)i * C * C4;
        const float* f2b = fc2_b + (size_t)i * C;

        layernorm_k<<<(T + 7) / 8, 256, 0, s>>>(X, g1, b1, LNB, T, C);
        gemm(LNB, qw, qb, 1, nullptr, BIGB, 1, T, C3, C, 0, s);
        dim3 ag(32 * nw * nw, heads);
        win_attn_k<<<ag, 256, 0, s>>>(BIGB, rp, ATNB, Hs, C, heads, shift);
        gemm(ATNB, pw, pb, 1, TMPF, nullptr, 0, T, C, C, 0, s);
        add_inplace_k<<<(unsigned)((TC + 255) / 256), 256, 0, s>>>(X, TMPF, TC);
        layernorm_k<<<(T + 7) / 8, 256, 0, s>>>(X, g2, b2, LNB, T, C);
        gemm(LNB, f1w, f1b, 1, nullptr, BIGB, 1, T, C4, C, 1, s);   // fused exact GELU -> bf16
        gemm(BIGB, f2w, f2b, 1, TMPF, nullptr, 0, T, C, C4, 0, s);
        add_inplace_k<<<(unsigned)((TC + 255) / 256), 256, 0, s>>>(X, TMPF, TC);
    }
}

static void patch_merge(const float* g, const float* b, const unsigned short* wB,
                        float* X, float* TMPF, unsigned short* LNB,
                        int H2, int Cin, hipStream_t s)
{
    int T2 = 32 * H2 * H2, C4 = 4 * Cin, Cout = 2 * Cin;
    size_t total = (size_t)T2 * C4;
    pm_gather_k<<<(unsigned)((total + 255) / 256), 256, 0, s>>>(X, TMPF, H2, Cin);
    layernorm_k<<<(T2 + 7) / 8, 256, 0, s>>>(TMPF, g, b, LNB, T2, C4);
    gemm(LNB, wB, nullptr, 0, X, nullptr, 0, T2, Cout, C4, 0, s);
}

extern "C" void kernel_launch(void* const* d_in, const int* in_sizes, int n_in,
                              void* d_out, int out_size, void* d_ws, size_t ws_size,
                              hipStream_t stream)
{
    (void)n_in; (void)out_size; (void)ws_size;

    // ---- workspace carve-up ----
    float* X    = (float*)d_ws;                                  // 12845056 f  residual
    float* TMPF = X + 12845056;                                  // 12845056 f  f32 scratch
    unsigned short* LNB  = (unsigned short*)(TMPF + 12845056);   // 12845056 u16
    unsigned short* ATNB = LNB + 12845056;                       // 12845056 u16
    unsigned short* BIGB = ATNB + 12845056;                      // 51380224 u16
    unsigned short* warena = BIGB + 51380224;                    // bf16 weights

    // ---- convert all GEMM weights to bf16 once per launch ----
    unsigned short* wp = warena;
    auto cvt = [&](int idx) -> unsigned short* {
        size_t n = (size_t)in_sizes[idx];
        unsigned short* dst = wp; wp += n;
        unsigned blocks = (unsigned)((n + 255) / 256);
        if (blocks > 8192) blocks = 8192;
        cvt_bf16_k<<<blocks, 256, 0, stream>>>((const float*)d_in[idx], dst, n);
        return dst;
    };
    unsigned short* s0_qkv = cvt(13); unsigned short* s0_prj = cvt(16);
    unsigned short* s0_f1  = cvt(20); unsigned short* s0_f2  = cvt(22);
    unsigned short* pm1_w  = cvt(26);
    unsigned short* s1_qkv = cvt(29); unsigned short* s1_prj = cvt(32);
    unsigned short* s1_f1  = cvt(36); unsigned short* s1_f2  = cvt(38);
    unsigned short* pm2_w  = cvt(42);
    unsigned short* s2_qkv = cvt(45); unsigned short* s2_prj = cvt(48);
    unsigned short* s2_f1  = cvt(52); unsigned short* s2_f2  = cvt(54);

    // ---- stem: conv3x3+BN+ReLU -> edge sigmoid gate -> 4x4 patch embed ----
    float* EDG = (float*)LNB;   // reuse (fully rewritten later)
    { int total = 32 * 3 * 224 * 224;
      conv1_bn_relu_k<<<(total + 255) / 256, 256, 0, stream>>>(
          (const float*)d_in[0], (const float*)d_in[1], (const float*)d_in[2],
          (const float*)d_in[3], (const float*)d_in[4], (const float*)d_in[5],
          (const float*)d_in[6], TMPF); }
    { int total = 32 * 224 * 224;
      edge_mul_k<<<(total + 255) / 256, 256, 0, stream>>>(
          TMPF, (const float*)d_in[7], (const float*)d_in[8], EDG); }
    { int total = 32 * 56 * 56 * 128;
      patch_embed_k<<<(total + 255) / 256, 256, 0, stream>>>(
          EDG, (const float*)d_in[9], (const float*)d_in[10], X); }

    // ---- stage 0: T=100352, H=56, C=128, heads=4, 2 blocks ----
    run_stage(d_in, 11, s0_qkv, s0_prj, s0_f1, s0_f2, X, TMPF, LNB, ATNB, BIGB,
              100352, 56, 128, 4, 2, stream);
    patch_merge((const float*)d_in[24], (const float*)d_in[25], pm1_w, X, TMPF, LNB, 28, 128, stream);

    // ---- stage 1: T=25088, H=28, C=256, heads=8, 2 blocks ----
    run_stage(d_in, 27, s1_qkv, s1_prj, s1_f1, s1_f2, X, TMPF, LNB, ATNB, BIGB,
              25088, 28, 256, 8, 2, stream);
    patch_merge((const float*)d_in[40], (const float*)d_in[41], pm2_w, X, TMPF, LNB, 14, 256, stream);

    // ---- stage 2: T=6272, H=14, C=512, heads=16, 6 blocks ----
    run_stage(d_in, 43, s2_qkv, s2_prj, s2_f1, s2_f2, X, TMPF, LNB, ATNB, BIGB,
              6272, 14, 512, 16, 6, stream);

    // ---- head: mean pool -> dense(512)+ELU -> dense(1) ----
    float* HID = (float*)ATNB;
    meanpool_k<<<(32 * 512 + 255) / 256, 256, 0, stream>>>(X, TMPF, 14, 512);
    dense_elu_k<<<(32 * 512 + 255) / 256, 256, 0, stream>>>(
        TMPF, (const float*)d_in[56], (const float*)d_in[57], HID, 512, 512);
    dense_out_k<<<1, 32, 0, stream>>>(HID, (const float*)d_in[58], (const float*)d_in[59], (float*)d_out);
}
